// NormEMAVectorQuantizer_1614907703803
// MI455X (gfx1250) — compile-verified, matrix operand order checked by reading, not run on hardware
//
#include <hip/hip_runtime.h>
#include <hip/hip_bf16.h>

typedef __attribute__((ext_vector_type(8)))  __bf16 v8bf;
typedef __attribute__((ext_vector_type(16))) __bf16 v16bf;
typedef __attribute__((ext_vector_type(8)))  float  v8f;

#define D 256            // feature dim (fixed by reference)
#define KSTEPS 8         // 256 / 32
#define LDS_ROW 264      // 264 bf16 = 528 B row stride (132 dwords -> 4-bank skew, conflict-free)
#define NSPLIT 2         // codebook split across gridDim.y

// ---------------- prep: normalize z rows, convert to bf16 ----------------
__global__ __launch_bounds__(256) void prep_z_kernel(const float* __restrict__ z,
                                                     __hip_bfloat16* __restrict__ znbf) {
    const int row = blockIdx.x;
    const int t   = threadIdx.x;
    float x = z[row * D + t];
    __shared__ float red[256];
    red[t] = x * x;
    __syncthreads();
    for (int s = 128; s > 0; s >>= 1) {
        if (t < s) red[t] += red[t + s];
        __syncthreads();
    }
    float nrm   = sqrtf(red[0]);
    float scale = 1.0f / fmaxf(nrm, 1e-12f);
    znbf[row * D + t] = __float2bfloat16(x * scale);
}

// ---------------- prep: weight -> bf16 + exact fp32 ||w||^2 ----------------
__global__ __launch_bounds__(256) void prep_w_kernel(const float* __restrict__ w,
                                                     __hip_bfloat16* __restrict__ wbf,
                                                     float* __restrict__ wsq) {
    const int row = blockIdx.x;
    const int t   = threadIdx.x;
    float x = w[row * D + t];
    __shared__ float red[256];
    red[t] = x * x;
    __syncthreads();
    for (int s = 128; s > 0; s >>= 1) {
        if (t < s) red[t] += red[t + s];
        __syncthreads();
    }
    if (t == 0) wsq[row] = red[0];
    wbf[row * D + t] = __float2bfloat16(x);
}

static __device__ __forceinline__ v16bf cat8(v8bf lo, v8bf hi) {
    return __builtin_shufflevector(lo, hi, 0,1,2,3,4,5,6,7,8,9,10,11,12,13,14,15);
}

// ---------------- main: WMMA distance + per-row partial argmin ----------------
// Block = 256 threads = 8 waves; wave w owns TWO 16-row A tiles (rows blk*256 + w*32 .. +32),
// both held in VGPRs, so each B tile read from LDS feeds two WMMAs (halves DS bytes/WMMA).
// gridDim.y = NSPLIT splits the codebook; partial (val,idx) merged in the gather kernel.
// B tiles staged into LDS via async global->LDS DMA, double-buffered.
__global__ __launch_bounds__(256, 1) void vq_argmin_kernel(
        const __hip_bfloat16* __restrict__ zn,
        const __hip_bfloat16* __restrict__ wb,
        const float* __restrict__ wsq,
        float* __restrict__ pval,       // [BT * NSPLIT]
        int*   __restrict__ pidx,       // [BT * NSPLIT]
        int N) {
    __shared__ __hip_bfloat16 ldsB[2][16 * LDS_ROW];

    const int lane  = threadIdx.x & 31;
    const int wave  = threadIdx.x >> 5;
    const int split = blockIdx.y;
    const int nbase = split * (N / NSPLIT);
    const int m0    = blockIdx.x * 256 + wave * 32;   // first of two 16-row tiles
    const int mr    = lane & 15;   // row (A) / col (B) within a 16-tile
    const int half  = lane >> 4;   // K-chunk select per ISA 16-bit A/B layout

    // --- preload A: two 16x256 bf16 tiles in ISA layout (stay in VGPRs) ---
    const __hip_bfloat16* arow0 = zn + (size_t)(m0 + mr)      * D + half * 8;
    const __hip_bfloat16* arow1 = zn + (size_t)(m0 + 16 + mr) * D + half * 8;
    v16bf a0[KSTEPS], a1[KSTEPS];
#pragma unroll
    for (int kk = 0; kk < KSTEPS; ++kk) {
        a0[kk] = cat8(*(const v8bf*)(const void*)(arow0 + kk * 32),
                      *(const v8bf*)(const void*)(arow0 + kk * 32 + 16));
        a1[kk] = cat8(*(const v8bf*)(const void*)(arow1 + kk * 32),
                      *(const v8bf*)(const void*)(arow1 + kk * 32 + 16));
    }

    // LDS byte offsets of the two staging buffers (low 32 bits of flat shared addr)
    const unsigned lb[2] = { (unsigned)(uintptr_t)&ldsB[0][0],
                             (unsigned)(uintptr_t)&ldsB[1][0] };

    // async DMA of one 16-row B tile (16 x 512B) into buffer `buf`
    // 512 x 16B chunks; 2 per thread; fully coalesced on the global side
    auto issue_tile = [&](int buf, int n0) {
#pragma unroll
        for (int j = 0; j < 2; ++j) {
            const int c   = threadIdx.x + 256 * j;   // chunk 0..511
            const int row = c >> 5;                  // 32 chunks per 512B row
            const int col = c & 31;
            unsigned ldsa = lb[buf] + row * (LDS_ROW * 2) + col * 16;
            const __hip_bfloat16* g = wb + (size_t)(n0 + row) * D + col * 8;
            asm volatile("global_load_async_to_lds_b128 %0, %1, off"
                         :: "v"(ldsa), "v"(g) : "memory");
        }
    };

    float bestv0[8], bestv1[8];
    int   besti0[8], besti1[8];
#pragma unroll
    for (int r = 0; r < 8; ++r) {
        bestv0[r] = 3.0e38f; besti0[r] = 0;
        bestv1[r] = 3.0e38f; besti1[r] = 0;
    }

    const int ntiles = (N / NSPLIT) >> 4;            // 16 codes per iteration
    issue_tile(0, nbase);

#pragma unroll 1
    for (int i = 0; i < ntiles; ++i) {
        const int cur = i & 1;
        const int n0  = nbase + (i << 4);
        if (i + 1 < ntiles) {
            issue_tile(cur ^ 1, n0 + 16);
            asm volatile("s_wait_asynccnt 0x2" ::: "memory");  // current tile's 2 DMAs done
        } else {
            asm volatile("s_wait_asynccnt 0x0" ::: "memory");
        }
        __syncthreads();                 // publish LDS writes to all waves

        const float wsqn = wsq[n0 + mr];
        const __hip_bfloat16* brow = &ldsB[cur][ (size_t)mr * LDS_ROW + half * 8];

        v8f acc0 = {};
        v8f acc1 = {};
#pragma unroll
        for (int kk = 0; kk < KSTEPS; ++kk) {
            v16bf b = cat8(*(const v8bf*)(const void*)(brow + kk * 32),
                           *(const v8bf*)(const void*)(brow + kk * 32 + 16));
            acc0 = __builtin_amdgcn_wmma_f32_16x16x32_bf16(
                       false, a0[kk], false, b, (short)0, acc0, false, false);
            acc1 = __builtin_amdgcn_wmma_f32_16x16x32_bf16(
                       false, a1[kk], false, b, (short)0, acc1, false, false);
        }

        const int n = n0 + mr;
#pragma unroll
        for (int r = 0; r < 8; ++r) {
            // d = ||w||^2 - 2*dot  (||zn||^2 = 1 is argmin-invariant)
            float d0 = __builtin_fmaf(-2.0f, acc0[r], wsqn);
            float d1 = __builtin_fmaf(-2.0f, acc1[r], wsqn);
            if (d0 < bestv0[r]) { bestv0[r] = d0; besti0[r] = n; }
            if (d1 < bestv1[r]) { bestv1[r] = d1; besti1[r] = n; }
        }
        __syncthreads();                 // all waves done reading buf before it is re-filled
    }

    // --- reduce (val, idx) across the 16-lane half; tie-break to lower idx ---
#pragma unroll
    for (int t = 0; t < 2; ++t) {
#pragma unroll
        for (int r = 0; r < 8; ++r) {
            float v  = t ? bestv1[r] : bestv0[r];
            int   ix = t ? besti1[r] : besti0[r];
#pragma unroll
            for (int off = 1; off < 16; off <<= 1) {
                float v2 = __shfl_xor(v, off, 16);
                int   i2 = __shfl_xor(ix, off, 16);
                if (v2 < v || (v2 == v && i2 < ix)) { v = v2; ix = i2; }
            }
            if (mr == 0) {
                const int m = m0 + t * 16 + half * 8 + r;   // row m (C/D layout)
                pval[(size_t)m * NSPLIT + split] = v;
                pidx[(size_t)m * NSPLIT + split] = ix;
            }
        }
    }
}

// ---------------- merge splits + gather: z_q = weight[idx] (exact fp32) ----------------
__global__ __launch_bounds__(256) void gather_kernel(const float* __restrict__ w,
                                                     const float* __restrict__ pval,
                                                     const int* __restrict__ pidx,
                                                     float* __restrict__ zq,
                                                     float* __restrict__ idxf) {
    const int m = blockIdx.x;
    const int t = threadIdx.x;
    const float v0 = pval[(size_t)m * NSPLIT + 0];
    const float v1 = pval[(size_t)m * NSPLIT + 1];
    const int   i0 = pidx[(size_t)m * NSPLIT + 0];
    const int   i1 = pidx[(size_t)m * NSPLIT + 1];
    // split 0 holds strictly lower code indices; strict '<' keeps argmin-first on ties
    const int ix = (v1 < v0) ? i1 : i0;
    zq[(size_t)m * D + t] = w[(size_t)ix * D + t];
    if (t == 0) idxf[m] = (float)ix;
}

extern "C" void kernel_launch(void* const* d_in, const int* in_sizes, int n_in,
                              void* d_out, int out_size, void* d_ws, size_t ws_size,
                              hipStream_t stream) {
    const float* z = (const float*)d_in[0];   // [B*T, 256] = [16384, 256] fp32
    const float* w = (const float*)d_in[1];   // [8192, 256] fp32
    const int BT = in_sizes[0] / D;           // 16384
    const int N  = in_sizes[1] / D;           // 8192

    // Workspace layout
    char* ws = (char*)d_ws;
    __hip_bfloat16* znbf = (__hip_bfloat16*)ws;                              // BT*D*2
    __hip_bfloat16* wbf  = (__hip_bfloat16*)(ws + (size_t)BT * D * 2);       // N*D*2
    float* wsq  = (float*)(ws + (size_t)BT * D * 2 + (size_t)N * D * 2);     // N*4
    float* pval = (float*)((char*)wsq + (size_t)N * 4);                      // BT*NSPLIT*4
    int*   pidx = (int*)((char*)pval + (size_t)BT * NSPLIT * 4);             // BT*NSPLIT*4

    float* zq   = (float*)d_out;              // BT*D fp32
    float* idxf = zq + (size_t)BT * D;        // BT (indices as exact fp32 values)

    prep_z_kernel<<<BT, 256, 0, stream>>>(z, znbf);
    prep_w_kernel<<<N, 256, 0, stream>>>(w, wbf, wsq);
    dim3 grid(BT / 256, NSPLIT);
    vq_argmin_kernel<<<grid, 256, 0, stream>>>(znbf, wbf, wsq, pval, pidx, N);
    gather_kernel<<<BT, 256, 0, stream>>>(w, pval, pidx, zq, idxf);
}
